// SS_MoGA_RNNAgent_69836168233091
// MI455X (gfx1250) — matrix operand order repaired and checked by reading, not compiled, not run on hardware
//
#include <hip/hip_runtime.h>
#include <hip/hip_bf16.h>
#include <math.h>

// ---------------------------------------------------------------------------
// SS_MoGA_RNNAgent forward for MI455X (gfx1250, wave32, WMMA).
// All large GEMMs -> v_wmma_f32_16x16x32_bf16 (bf16 in, f32 accumulate).
// FAST GEMM path: double-buffered LDS (compile-time buffer index),
// HBM->LDS via global_load_async_to_lds_b128 (ASYNCcnt), software-pipelined.
// Fragments loaded as explicit bf16x8 vector LDS loads (ds_load_b128).
// ---------------------------------------------------------------------------

typedef __attribute__((ext_vector_type(16))) __bf16 bf16x16;
typedef __attribute__((ext_vector_type(8)))  __bf16 bf16x8;
typedef __attribute__((ext_vector_type(8)))  float  f32x8;

#define BN   16384
#define NAL  15
#define NEN  16
#define HDIM 256
#define NHD  4
#define DHD  64

#define XS_STRIDE 40   // bf16; 80B pitch: 16B-aligned chunks, conflict-free gathers

// One K-step of WMMAs for a wave: 2(M) x 4(N) subtiles of 16x16 from LDS tiles.
__device__ __forceinline__ void wmma_tile(
    const __bf16* __restrict__ xs, const __bf16* __restrict__ ws,
    int wm, int wn, int g, int l16, f32x8 (&acc)[2][4])
{
    bf16x16 fa[2];
    #pragma unroll
    for (int ms = 0; ms < 2; ++ms) {
        const __bf16* row = xs + (wm * 32 + ms * 16 + l16) * XS_STRIDE;
        bf16x8 lo = *(const bf16x8*)(row + g * 8);        // K 0..7  (g=0) / 8..15  (g=1)
        bf16x8 hi = *(const bf16x8*)(row + 16 + g * 8);   // K 16..23 (g=0) / 24..31 (g=1)
        fa[ms] = __builtin_shufflevector(lo, hi, 0,1,2,3,4,5,6,7,8,9,10,11,12,13,14,15);
    }
    #pragma unroll
    for (int ns = 0; ns < 4; ++ns) {
        const __bf16* row = ws + (wn * 64 + ns * 16 + l16) * XS_STRIDE;
        bf16x8 lo = *(const bf16x8*)(row + g * 8);        // K = 8g + 0..7
        bf16x8 hi = *(const bf16x8*)(row + 16 + g * 8);   // K = 16 + 8g + 0..7
        bf16x16 fb = __builtin_shufflevector(lo, hi, 0,1,2,3,4,5,6,7,8,9,10,11,12,13,14,15);
        acc[0][ns] = __builtin_amdgcn_wmma_f32_16x16x32_bf16(
            false, fa[0], false, fb, (short)0, acc[0][ns], false, false);
        acc[1][ns] = __builtin_amdgcn_wmma_f32_16x16x32_bf16(
            false, fa[1], false, fb, (short)0, acc[1][ns], false, false);
    }
}

// ---------------------------------------------------------------------------
// Tiled GEMM:  Y[M,N] = X[M,K] @ W[N,K]^T + bias[N]   (X, W bf16; bias != null)
// Block tile 128x128, 8 waves in 4(M) x 2(N).
// FAST=true requires M%128==0, N%128==0, K%32==0.
// ---------------------------------------------------------------------------
template <typename YT, bool FAST>
__global__ __launch_bounds__(256) void gemm_xwt_wmma(
    const __bf16* __restrict__ X, const __bf16* __restrict__ W,
    const float* __restrict__ bias, YT* __restrict__ Y,
    int M, int N, int K)
{
    __shared__ __bf16 Xs[2][128][XS_STRIDE];
    __shared__ __bf16 Ws[2][128][XS_STRIDE];

    const int tid  = threadIdx.x;
    const int lane = tid & 31;
    const int wid  = tid >> 5;
    const int wm   = wid & 3;        // wave row 0..3  (32 rows each)
    const int wn   = wid >> 2;       // wave col 0..1  (64 cols each)
    const int g    = lane >> 4;      // lane group 0/1
    const int l16  = lane & 15;

    const int m_blk = blockIdx.x * 128;
    const int n_blk = blockIdx.y * 128;

    f32x8 acc[2][4];
    for (int i = 0; i < 2; ++i)
        for (int j = 0; j < 4; ++j)
            for (int r = 0; r < 8; ++r) acc[i][j][r] = 0.0f;

    if (FAST) {
        // per-thread fixed (row, chunk) assignment: 2 chunks of 16B each
        const int r0 = tid >> 2, c0 = (tid & 3) * 8;      // chunk 0: rows 0..63
        const int r1 = r0 + 64;                           // chunk 1: rows 64..127
        const __bf16* gx0 = X + (size_t)(m_blk + r0) * K + c0;
        const __bf16* gx1 = X + (size_t)(m_blk + r1) * K + c0;
        const __bf16* gw0 = W + (size_t)(n_blk + r0) * K + c0;
        const __bf16* gw1 = W + (size_t)(n_blk + r1) * K + c0;

        const int nk = K >> 5;

        auto issue = [&](int i, int buf) {
            unsigned lx0 = (unsigned)(size_t)&Xs[buf][r0][c0];
            unsigned lx1 = (unsigned)(size_t)&Xs[buf][r1][c0];
            unsigned lw0 = (unsigned)(size_t)&Ws[buf][r0][c0];
            unsigned lw1 = (unsigned)(size_t)&Ws[buf][r1][c0];
            const __bf16* a0 = gx0 + (i << 5);
            const __bf16* a1 = gx1 + (i << 5);
            const __bf16* b0 = gw0 + (i << 5);
            const __bf16* b1 = gw1 + (i << 5);
            asm volatile("global_load_async_to_lds_b128 %0, %1, off"
                         :: "v"(lx0), "v"(a0) : "memory");
            asm volatile("global_load_async_to_lds_b128 %0, %1, off"
                         :: "v"(lx1), "v"(a1) : "memory");
            asm volatile("global_load_async_to_lds_b128 %0, %1, off"
                         :: "v"(lw0), "v"(b0) : "memory");
            asm volatile("global_load_async_to_lds_b128 %0, %1, off"
                         :: "v"(lw1), "v"(b1) : "memory");
        };

        issue(0, 0);
        // 2x-unrolled pipeline so the LDS buffer index is a compile-time constant.
        for (int i = 0; i < nk; i += 2) {
            // even step: compute from buffer 0
            if (i + 1 < nk) {
                issue(i + 1, 1);
                // 4 async instrs per wave per tile; in-order completion:
                // <=4 outstanding means the current tile has fully landed.
                asm volatile("s_wait_asynccnt 0x4" ::: "memory");
            } else {
                asm volatile("s_wait_asynccnt 0x0" ::: "memory");
            }
            __syncthreads();
            wmma_tile(&Xs[0][0][0], &Ws[0][0][0], wm, wn, g, l16, acc);
            __syncthreads();
            if (i + 1 >= nk) break;

            // odd step: compute from buffer 1
            if (i + 2 < nk) {
                issue(i + 2, 0);
                asm volatile("s_wait_asynccnt 0x4" ::: "memory");
            } else {
                asm volatile("s_wait_asynccnt 0x0" ::: "memory");
            }
            __syncthreads();
            wmma_tile(&Xs[1][0][0], &Ws[1][0][0], wm, wn, g, l16, acc);
            __syncthreads();
        }

        // unguarded epilogue (bias always present)
        #pragma unroll
        for (int ms = 0; ms < 2; ++ms)
            #pragma unroll
            for (int ns = 0; ns < 4; ++ns) {
                const int gn  = n_blk + wn * 64 + ns * 16 + l16;
                const int gm0 = m_blk + wm * 32 + ms * 16 + g * 8;
                const float bv = bias[gn];
                YT* yp = Y + (size_t)gm0 * N + gn;
                #pragma unroll
                for (int r = 0; r < 8; ++r) {
                    yp[0] = (YT)(acc[ms][ns][r] + bv);
                    yp += N;
                }
            }
    } else {
        // guarded scalar path (K tails, e.g. K=30 own embedding)
        for (int k0 = 0; k0 < K; k0 += 32) {
            for (int e = tid; e < 128 * 32; e += 256) {
                int r = e >> 5, k = e & 31;
                int gm = m_blk + r, gk = k0 + k;
                __bf16 v = (__bf16)0.0f;
                if (gm < M && gk < K) v = X[(long long)gm * K + gk];
                Xs[0][r][k] = v;
            }
            for (int e = tid; e < 128 * 32; e += 256) {
                int r = e >> 5, k = e & 31;
                int gn = n_blk + r, gk = k0 + k;
                __bf16 v = (__bf16)0.0f;
                if (gn < N && gk < K) v = W[(long long)gn * K + gk];
                Ws[0][r][k] = v;
            }
            __syncthreads();
            wmma_tile(&Xs[0][0][0], &Ws[0][0][0], wm, wn, g, l16, acc);
            __syncthreads();
        }

        for (int ms = 0; ms < 2; ++ms)
            for (int ns = 0; ns < 4; ++ns) {
                int gn = n_blk + wn * 64 + ns * 16 + l16;
                #pragma unroll
                for (int r = 0; r < 8; ++r) {
                    int gm = m_blk + wm * 32 + ms * 16 + g * 8 + r;
                    if (gm < M && gn < N)
                        Y[(long long)gm * N + gn] = (YT)(acc[ms][ns][r] + bias[gn]);
                }
            }
    }
}

template <typename YT>
static void run_gemm(const __bf16* X, const __bf16* W, const float* bias, YT* Y,
                     int M, int N, int K, hipStream_t s)
{
    dim3 grid((M + 127) / 128, (N + 127) / 128);
    const bool fast = (M % 128 == 0) && (N % 128 == 0) && (K % 32 == 0);
    if (fast)
        gemm_xwt_wmma<YT, true ><<<grid, 256, 0, s>>>(X, W, bias, Y, M, N, K);
    else
        gemm_xwt_wmma<YT, false><<<grid, 256, 0, s>>>(X, W, bias, Y, M, N, K);
}

// ---------------------------------------------------------------------------
// f32 -> bf16 conversion (weights once, activations where needed)
// ---------------------------------------------------------------------------
__global__ void cvt_f32_bf16(const float* __restrict__ s, __bf16* __restrict__ d,
                             long long n)
{
    long long i = (long long)blockIdx.x * blockDim.x + threadIdx.x;
    if (i < n) d[i] = (__bf16)s[i];
}
static void run_cvt(const float* s, __bf16* d, long long n, hipStream_t st)
{
    cvt_f32_bf16<<<(unsigned)((n + 255) / 256), 256, 0, st>>>(s, d, n);
}

// ---------------------------------------------------------------------------
// Masks + group temperatures
// ---------------------------------------------------------------------------
__device__ __forceinline__ float softplusf(float x)
{
    return x > 20.0f ? x : log1pf(expf(x));
}

__global__ void prep_kernel(const float* __restrict__ ally,
                            const float* __restrict__ enemy,
                            const float* __restrict__ gtw,
                            const float* __restrict__ gtb,
                            unsigned* __restrict__ mA, unsigned* __restrict__ mE,
                            float* __restrict__ tA, float* __restrict__ tE)
{
    int b = blockIdx.x * blockDim.x + threadIdx.x;
    if (b >= BN) return;
    unsigned ma = 0;
    for (int n = 0; n < NAL; ++n) {
        const float* p = ally + ((long long)b * NAL + n) * 32;
        bool any = false;
        for (int d = 0; d < 32; ++d) any |= (p[d] != 0.0f);
        if (any) ma |= (1u << n);
    }
    unsigned me = 0;
    for (int n = 0; n < NEN; ++n) {
        const float* p = enemy + ((long long)b * NEN + n) * 32;
        bool any = false;
        for (int d = 0; d < 32; ++d) any |= (p[d] != 0.0f);
        if (any) me |= (1u << n);
    }
    mA[b] = ma; mE[b] = me;
    float ca = (float)__popc(ma), ce = (float)__popc(me);
    tA[b] = softplusf(gtw[0] * log1pf(ca) + gtb[0]) + 1.0f;
    tE[b] = softplusf(gtw[1] * log1pf(ce) + gtb[1]) + 1.0f;
}

// ---------------------------------------------------------------------------
// Attention combine: softmax over <=16 entities per (batch, head), wave32.
// Block = 128 threads = 4 waves = 4 heads of one batch row. zcat out bf16.
// ---------------------------------------------------------------------------
__global__ __launch_bounds__(128) void attn_combine_kernel(
    const float* __restrict__ qh, const __bf16* __restrict__ kh,
    const __bf16* __restrict__ vh, const unsigned* __restrict__ maskbits,
    const float* __restrict__ temp, __bf16* __restrict__ zcat, int N)
{
    __shared__ float qsh[256];
    const int b = blockIdx.x;
    const int t = threadIdx.x;
    qsh[t]       = qh[(long long)b * 256 + t];
    qsh[t + 128] = qh[(long long)b * 256 + t + 128];
    __syncthreads();

    const int h = t >> 5, lane = t & 31;
    const unsigned mb = maskbits[b];
    const float tmp = temp[b];

    float s = -1e30f;
    const bool act = (lane < N) && ((mb >> lane) & 1u);
    if (act) {
        const __bf16* kr = kh + (long long)(b * N + lane) * 256 + h * 64;
        float d = 0.0f;
        for (int i = 0; i < DHD; ++i) d += qsh[h * 64 + i] * (float)kr[i];
        s = d * 0.125f / tmp;   // / sqrt(64) / temp
    }
    float mx = s;
    for (int off = 16; off > 0; off >>= 1) mx = fmaxf(mx, __shfl_xor(mx, off, 32));
    float e = act ? __expf(s - mx) : 0.0f;
    float sum = e;
    for (int off = 16; off > 0; off >>= 1) sum += __shfl_xor(sum, off, 32);
    float p = e / fmaxf(sum, 1e-30f);

    float z0 = 0.0f, z1 = 0.0f;
    for (int n = 0; n < N; ++n) {
        float pn = __shfl(p, n, 32);
        const __bf16* vr = vh + (long long)(b * N + n) * 256 + h * 64;
        z0 += pn * (float)vr[lane];
        z1 += pn * (float)vr[lane + 32];
    }
    zcat[(long long)b * 256 + h * 64 + lane]      = (__bf16)z0;
    zcat[(long long)b * 256 + h * 64 + lane + 32] = (__bf16)z1;
}

// ---------------------------------------------------------------------------
// Block-wide sum over 256 threads
// ---------------------------------------------------------------------------
__device__ __forceinline__ float block_sum256(float v, float* red)
{
    int t = threadIdx.x;
    red[t] = v; __syncthreads();
    for (int s = 128; s > 0; s >>= 1) {
        if (t < s) red[t] += red[t + s];
        __syncthreads();
    }
    float r = red[0]; __syncthreads();
    return r;
}

// ---------------------------------------------------------------------------
// CLN(A), CLN(E), then LN over concat(q_src, zA, zE) -> u (bf16, Bn x 768)
// ---------------------------------------------------------------------------
__global__ __launch_bounds__(256) void cln_u_kernel(
    const float* __restrict__ zApre, const float* __restrict__ zEpre,
    const float* __restrict__ gbA,   const float* __restrict__ gbE,
    const float* __restrict__ qsrc,
    const float* __restrict__ g3, const float* __restrict__ b3,
    __bf16* __restrict__ u)
{
    __shared__ float red[256];
    const int b = blockIdx.x, t = threadIdx.x;

    float xA  = zApre[(long long)b * 256 + t];
    float muA = block_sum256(xA, red) * (1.0f / 256.0f);
    float dA  = xA - muA;
    float vA  = block_sum256(dA * dA, red) * (1.0f / 256.0f);
    float zA  = (1.0f + gbA[(long long)b * 512 + t]) * (dA * rsqrtf(vA + 1e-5f))
              + gbA[(long long)b * 512 + 256 + t];

    float xE  = zEpre[(long long)b * 256 + t];
    float muE = block_sum256(xE, red) * (1.0f / 256.0f);
    float dE  = xE - muE;
    float vE  = block_sum256(dE * dE, red) * (1.0f / 256.0f);
    float zE  = (1.0f + gbE[(long long)b * 512 + t]) * (dE * rsqrtf(vE + 1e-5f))
              + gbE[(long long)b * 512 + 256 + t];

    float q   = qsrc[(long long)b * 256 + t];
    float su  = block_sum256(q + zA + zE, red);
    float sq  = block_sum256(q * q + zA * zA + zE * zE, red);
    float mu  = su * (1.0f / 768.0f);
    float var = sq * (1.0f / 768.0f) - mu * mu;
    float r   = rsqrtf(var + 1e-5f);

    u[(long long)b * 768 + t]       = (__bf16)((q  - mu) * r * g3[t]       + b3[t]);
    u[(long long)b * 768 + 256 + t] = (__bf16)((zA - mu) * r * g3[256 + t] + b3[256 + t]);
    u[(long long)b * 768 + 512 + t] = (__bf16)((zE - mu) * r * g3[512 + t] + b3[512 + t]);
}

// ---------------------------------------------------------------------------
// GRU gates; writes f32 h (model output) and bf16 h (for head GEMMs)
// ---------------------------------------------------------------------------
__global__ void gru_kernel(const float* __restrict__ gi, const float* __restrict__ gh,
                           const float* __restrict__ hprev,
                           float* __restrict__ hout, __bf16* __restrict__ hbf)
{
    long long idx = (long long)blockIdx.x * blockDim.x + threadIdx.x;
    if (idx >= (long long)BN * 256) return;
    long long b = idx >> 8;
    int t = (int)(idx & 255);
    const float* gib = gi + b * 768;
    const float* ghb = gh + b * 768;
    float r  = 1.0f / (1.0f + expf(-(gib[t]       + ghb[t])));
    float zg = 1.0f / (1.0f + expf(-(gib[256 + t] + ghb[256 + t])));
    float n  = tanhf(gib[512 + t] + r * ghb[512 + t]);
    float hp = hprev[idx];
    float h  = (1.0f - zg) * n + zg * hp;
    hout[idx] = h;
    hbf[idx]  = (__bf16)h;
}

// ---------------------------------------------------------------------------
// Heads: V, tb, Q_move (prototypes), Q_shoot (Ke attention logits)
// ---------------------------------------------------------------------------
__global__ __launch_bounds__(256) void heads_kernel(
    const float* __restrict__ h,   const float* __restrict__ qmv,
    const float* __restrict__ qsv, const __bf16* __restrict__ Ke,
    const float* __restrict__ Wv,  const float* __restrict__ bv,
    const float* __restrict__ Wtb, const float* __restrict__ btb,
    const float* __restrict__ log_tau, const float* __restrict__ prot,
    const unsigned* __restrict__ mE, float* __restrict__ Q)
{
    __shared__ float red[256];
    const int b = blockIdx.x, t = threadIdx.x;
    float hv = h[(long long)b * 256 + t];
    float qm = qmv[(long long)b * 256 + t];
    float qs = qsv[(long long)b * 256 + t];

    float oV  = block_sum256(hv * Wv[t],        red);
    float ot0 = block_sum256(hv * Wtb[t],       red);
    float ot1 = block_sum256(hv * Wtb[256 + t], red);
    float oA[6];
    for (int a = 0; a < 6; ++a)
        oA[a] = block_sum256(qm * prot[a * 256 + t], red);
    float oK[16];
    for (int n = 0; n < NEN; ++n)
        oK[n] = block_sum256(qs * (float)Ke[((long long)b * NEN + n) * 256 + t], red);

    if (t == 0) {
        float V      = oV  + bv[0];
        float bmove  = ot0 + btb[0];
        float bshoot = ot1 + btb[1];
        float tauM = expf(log_tau[0]) + 1e-6f;
        float tauS = expf(log_tau[1]) + 1e-6f;

        float Am[6], mean = 0.0f;
        for (int a = 0; a < 6; ++a) { Am[a] = oA[a] * (1.0f / 16.0f); mean += Am[a]; }
        mean *= (1.0f / 6.0f);
        for (int a = 0; a < 6; ++a)
            Q[(long long)b * 22 + a] = V + bmove + (Am[a] - mean) / tauM;

        unsigned mb = mE[b];
        float cnt = (float)__popc(mb & 0xFFFFu);
        float den = fmaxf(cnt, 1.0f);
        float an[16], msum = 0.0f;
        for (int n = 0; n < NEN; ++n) {
            an[n] = oK[n] * (1.0f / 16.0f);
            if ((mb >> n) & 1u) msum += an[n];
        }
        float mm = msum / den;
        for (int n = 0; n < NEN; ++n) {
            float av = ((mb >> n) & 1u) ? (an[n] - mm) : -1e30f;
            Q[(long long)b * 22 + 6 + n] = V + bshoot + av / tauS;
        }
    }
}

// ---------------------------------------------------------------------------
// Host orchestration
// ---------------------------------------------------------------------------
extern "C" void kernel_launch(void* const* d_in, const int* in_sizes, int n_in,
                              void* d_out, int out_size, void* d_ws, size_t ws_size,
                              hipStream_t stream)
{
    const float* own_raw   = (const float*)d_in[0];
    const float* ally_raw  = (const float*)d_in[1];
    const float* enemy_raw = (const float*)d_in[2];
    const float* hidden    = (const float*)d_in[3];
    const float* W_own     = (const float*)d_in[4];
    const float* b_own     = (const float*)d_in[5];
    const float* W_ally    = (const float*)d_in[6];
    const float* b_ally    = (const float*)d_in[7];
    const float* W_enemy   = (const float*)d_in[8];
    const float* b_enemy   = (const float*)d_in[9];
    const float* gt_w      = (const float*)d_in[10];
    const float* gt_b      = (const float*)d_in[11];
    const float* attnA_W   = (const float*)d_in[12];
    const float* attnA_b   = (const float*)d_in[13];
    const float* attnE_W   = (const float*)d_in[14];
    const float* attnE_b   = (const float*)d_in[15];
    const float* clnA_W    = (const float*)d_in[16];
    const float* clnA_b    = (const float*)d_in[17];
    const float* clnE_W    = (const float*)d_in[18];
    const float* clnE_b    = (const float*)d_in[19];
    const float* ln3_g     = (const float*)d_in[20];
    const float* ln3_b     = (const float*)d_in[21];
    const float* gru_Wih   = (const float*)d_in[22];
    const float* gru_Whh   = (const float*)d_in[23];
    const float* gru_bih   = (const float*)d_in[24];
    const float* gru_bhh   = (const float*)d_in[25];
    const float* Wv_head   = (const float*)d_in[26];
    const float* bv_head   = (const float*)d_in[27];
    const float* Wtb       = (const float*)d_in[28];
    const float* btb       = (const float*)d_in[29];
    const float* log_tau   = (const float*)d_in[30];
    const float* Wmq       = (const float*)d_in[31];
    const float* bmq       = (const float*)d_in[32];
    const float* prot      = (const float*)d_in[33];
    const float* Wqs       = (const float*)d_in[34];
    const float* bqs       = (const float*)d_in[35];
    const float* Wke       = (const float*)d_in[36];
    const float* bke       = (const float*)d_in[37];
    (void)in_sizes; (void)n_in; (void)out_size; (void)ws_size;

    char* base = (char*)d_ws;
    size_t off = 0;
    auto carve = [&](size_t bytes) -> void* {
        void* p = base + off;
        off = (off + bytes + 255) & ~(size_t)255;
        return p;
    };
    const int HH = HDIM * HDIM;   // 65536

    unsigned* mA    = (unsigned*)carve((size_t)BN * 4);
    unsigned* mEb   = (unsigned*)carve((size_t)BN * 4);
    float*    tA    = (float*)carve((size_t)BN * 4);
    float*    tE    = (float*)carve((size_t)BN * 4);
    // bf16 copies of raw inputs + hidden
    __bf16*   own_b   = (__bf16*)carve((size_t)BN * 30 * 2);
    __bf16*   ally_b  = (__bf16*)carve((size_t)BN * NAL * 32 * 2);
    __bf16*   enemy_b = (__bf16*)carve((size_t)BN * NEN * 32 * 2);
    __bf16*   hid_b   = (__bf16*)carve((size_t)BN * 256 * 2);
    // bf16 weights
    __bf16* Wown_b  = (__bf16*)carve((size_t)256 * 30 * 2);
    __bf16* Wally_b = (__bf16*)carve((size_t)256 * 32 * 2);
    __bf16* Wene_b  = (__bf16*)carve((size_t)256 * 32 * 2);
    __bf16* aAW_b   = (__bf16*)carve((size_t)4 * HH * 2);
    __bf16* aEW_b   = (__bf16*)carve((size_t)4 * HH * 2);
    __bf16* cAW_b   = (__bf16*)carve((size_t)2 * HH * 2);
    __bf16* cEW_b   = (__bf16*)carve((size_t)2 * HH * 2);
    __bf16* Wih_b   = (__bf16*)carve((size_t)768 * 768 * 2);
    __bf16* Whh_b   = (__bf16*)carve((size_t)768 * 256 * 2);
    __bf16* Wmq_b   = (__bf16*)carve((size_t)HH * 2);
    __bf16* Wqs_b   = (__bf16*)carve((size_t)HH * 2);
    __bf16* Wke_b   = (__bf16*)carve((size_t)HH * 2);
    // activations
    float*  qsrc   = (float*)carve((size_t)BN * 256 * 4);
    __bf16* qsrc_b = (__bf16*)carve((size_t)BN * 256 * 2);
    __bf16* Abf    = (__bf16*)carve((size_t)BN * NAL * 256 * 2);
    __bf16* Ebf    = (__bf16*)carve((size_t)BN * NEN * 256 * 2);
    float*  qhA    = (float*)carve((size_t)BN * 256 * 4);
    float*  qhE    = (float*)carve((size_t)BN * 256 * 4);
    __bf16* khb    = (__bf16*)carve((size_t)BN * NEN * 256 * 2);  // reused A then E
    __bf16* vhb    = (__bf16*)carve((size_t)BN * NEN * 256 * 2);
    __bf16* zcatA  = (__bf16*)carve((size_t)BN * 256 * 2);
    __bf16* zcatE  = (__bf16*)carve((size_t)BN * 256 * 2);
    float*  zApre  = (float*)carve((size_t)BN * 256 * 4);
    float*  zEpre  = (float*)carve((size_t)BN * 256 * 4);
    float*  gbA    = (float*)carve((size_t)BN * 512 * 4);
    float*  gbE    = (float*)carve((size_t)BN * 512 * 4);
    __bf16* u      = (__bf16*)carve((size_t)BN * 768 * 2);
    float*  gi     = (float*)carve((size_t)BN * 768 * 4);
    float*  gh     = (float*)carve((size_t)BN * 768 * 4);
    __bf16* h_b    = (__bf16*)carve((size_t)BN * 256 * 2);
    float*  qmv    = (float*)carve((size_t)BN * 256 * 4);
    float*  qsv    = (float*)carve((size_t)BN * 256 * 4);
    __bf16* Ke     = (__bf16*)carve((size_t)BN * NEN * 256 * 2);

    float* Qout = (float*)d_out;                  // (Bn, 22)
    float* hout = Qout + (size_t)BN * 22;         // (Bn, 256)

    // 0) conversions to bf16 (weights once; raw inputs)
    run_cvt(W_own,   Wown_b,  (long long)256 * 30, stream);
    run_cvt(W_ally,  Wally_b, (long long)256 * 32, stream);
    run_cvt(W_enemy, Wene_b,  (long long)256 * 32, stream);
    run_cvt(attnA_W, aAW_b,   (long long)4 * HH,   stream);
    run_cvt(attnE_W, aEW_b,   (long long)4 * HH,   stream);
    run_cvt(clnA_W,  cAW_b,   (long long)2 * HH,   stream);
    run_cvt(clnE_W,  cEW_b,   (long long)2 * HH,   stream);
    run_cvt(gru_Wih, Wih_b,   (long long)768 * 768, stream);
    run_cvt(gru_Whh, Whh_b,   (long long)768 * 256, stream);
    run_cvt(Wmq,     Wmq_b,   (long long)HH, stream);
    run_cvt(Wqs,     Wqs_b,   (long long)HH, stream);
    run_cvt(Wke,     Wke_b,   (long long)HH, stream);
    run_cvt(own_raw,   own_b,   (long long)BN * 30,       stream);
    run_cvt(ally_raw,  ally_b,  (long long)BN * NAL * 32, stream);
    run_cvt(enemy_raw, enemy_b, (long long)BN * NEN * 32, stream);
    run_cvt(hidden,    hid_b,   (long long)BN * 256,      stream);

    // 1) masks + temperatures
    prep_kernel<<<(BN + 255) / 256, 256, 0, stream>>>(
        ally_raw, enemy_raw, gt_w, gt_b, mA, mEb, tA, tE);

    // 2) embeddings
    run_gemm<float >(own_b,   Wown_b,  b_own,   qsrc, BN,       256, 30, stream);
    run_cvt(qsrc, qsrc_b, (long long)BN * 256, stream);
    run_gemm<__bf16>(ally_b,  Wally_b, b_ally,  Abf,  BN * NAL, 256, 32, stream);
    run_gemm<__bf16>(enemy_b, Wene_b,  b_enemy, Ebf,  BN * NEN, 256, 32, stream);

    // 3) ally attention
    run_gemm<float >(qsrc_b, aAW_b,          attnA_b,       qhA, BN,       256, 256, stream);
    run_gemm<__bf16>(Abf,    aAW_b + HH,     attnA_b + 256, khb, BN * NAL, 256, 256, stream);
    run_gemm<__bf16>(Abf,    aAW_b + 2 * HH, attnA_b + 512, vhb, BN * NAL, 256, 256, stream);
    attn_combine_kernel<<<BN, 128, 0, stream>>>(qhA, khb, vhb, mA, tA, zcatA, NAL);
    run_gemm<float>(zcatA, aAW_b + 3 * HH, attnA_b + 768, zApre, BN, 256, 256, stream);

    // 4) enemy attention (reuses kh/vh scratch)
    run_gemm<float >(qsrc_b, aEW_b,          attnE_b,       qhE, BN,       256, 256, stream);
    run_gemm<__bf16>(Ebf,    aEW_b + HH,     attnE_b + 256, khb, BN * NEN, 256, 256, stream);
    run_gemm<__bf16>(Ebf,    aEW_b + 2 * HH, attnE_b + 512, vhb, BN * NEN, 256, 256, stream);
    attn_combine_kernel<<<BN, 128, 0, stream>>>(qhE, khb, vhb, mEb, tE, zcatE, NEN);
    run_gemm<float>(zcatE, aEW_b + 3 * HH, attnE_b + 768, zEpre, BN, 256, 256, stream);

    // 5) conditioned LN gammas/betas (2H fused as N=512), then CLN + LN3 -> u
    run_gemm<float>(qsrc_b, cAW_b, clnA_b, gbA, BN, 512, 256, stream);
    run_gemm<float>(qsrc_b, cEW_b, clnE_b, gbE, BN, 512, 256, stream);
    cln_u_kernel<<<BN, 256, 0, stream>>>(zApre, zEpre, gbA, gbE, qsrc, ln3_g, ln3_b, u);

    // 6) GRU
    run_gemm<float>(u,     Wih_b, gru_bih, gi, BN, 768, 768, stream);
    run_gemm<float>(hid_b, Whh_b, gru_bhh, gh, BN, 768, 256, stream);
    gru_kernel<<<((long long)BN * 256 + 255) / 256, 256, 0, stream>>>(
        gi, gh, hidden, hout, h_b);

    // 7) heads
    run_gemm<float >(h_b, Wmq_b, bmq, qmv, BN,       256, 256, stream);
    run_gemm<float >(h_b, Wqs_b, bqs, qsv, BN,       256, 256, stream);
    run_gemm<__bf16>(Ebf, Wke_b, bke, Ke,  BN * NEN, 256, 256, stream);
    heads_kernel<<<BN, 256, 0, stream>>>(
        hout, qmv, qsv, Ke, Wv_head, bv_head, Wtb, btb, log_tau, prot, mEb, Qout);
}
